// CostVolumeEncoder_31104153158145
// MI455X (gfx1250) — matrix operand order repaired and checked by reading, not compiled, not run on hardware
//
#include <hip/hip_runtime.h>
#include <hip/hip_bf16.h>

typedef __attribute__((ext_vector_type(16))) __bf16 v16bf;
typedef __attribute__((ext_vector_type(8)))  __bf16 v8bf;
typedef __attribute__((ext_vector_type(8)))  float  v8f;
typedef __attribute__((ext_vector_type(4)))  unsigned int u32x4;
typedef __attribute__((ext_vector_type(8)))  int i32x8;
typedef __attribute__((ext_vector_type(4)))  int i32x4;

#define BN_SCALE 0.9999950000374997f  // 1/sqrt(1+1e-5)

// ---------------------------------------------------------------------------
// Weight conversion fp32 -> bf16 with zero padding to (Mpad x Kpad)
// ---------------------------------------------------------------------------
__global__ void k_cvt_pad_bf16(const float* __restrict__ src, __bf16* __restrict__ dst,
                               int M, int K, int Kp, long total) {
  long i = (long)blockIdx.x * blockDim.x + threadIdx.x;
  if (i >= total) return;
  int k = (int)(i % Kp);
  int m = (int)(i / Kp);
  float v = (m < M && k < K) ? src[(long)m * K + k] : 0.f;
  dst[i] = (__bf16)v;
}

// w: (Ci, Co, 3, 3) fp32 -> dst: (Mpad=pad(Co), Kpad=pad(Ci*9)) bf16
// with wt[o, i*9+ky*3+kx] = w[i, o, 2-ky, 2-kx], zero padded.
__global__ void k_cvt_deconv_pad_bf16(const float* __restrict__ w, __bf16* __restrict__ dst,
                                      int Ci, int Co, int Kp, long total) {
  long i = (long)blockIdx.x * blockDim.x + threadIdx.x;
  if (i >= total) return;
  int kk = (int)(i % Kp);
  int o  = (int)(i / Kp);
  float v = 0.f;
  if (o < Co && kk < Ci * 9) {
    int c = kk / 9;
    int rr = kk - c * 9;
    int ky = rr / 3;
    int kx = rr - ky * 3;
    v = w[((long)(c * Co + o) * 3 + (2 - ky)) * 3 + (2 - kx)];
  }
  dst[i] = (__bf16)v;
}

// ---------------------------------------------------------------------------
// SE path (tiny): bn_eval -> fc1+relu -> fc2 -> 1x1 relu -> 1x1 -> sigmoid
// ---------------------------------------------------------------------------
__global__ __launch_bounds__(128)
void k_se(const float* __restrict__ mlp, const float* __restrict__ bn_g,
          const float* __restrict__ bn_b, const float* __restrict__ fc1w,
          const float* __restrict__ fc1b, const float* __restrict__ fc2w,
          const float* __restrict__ fc2b, const float* __restrict__ srw,
          const float* __restrict__ srb, const float* __restrict__ sew,
          const float* __restrict__ seb, float* __restrict__ sig) {
  const int img = blockIdx.x;
  const int j = threadIdx.x;
  __shared__ float mi[32], h1[128], h2[128], s1[128];
  if (j < 27) mi[j] = mlp[img * 27 + j] * (bn_g[j] * BN_SCALE) + bn_b[j];
  __syncthreads();
  float a = fc1b[j];
  #pragma unroll 1
  for (int k = 0; k < 27; ++k) a += mi[k] * fc1w[j * 27 + k];
  h1[j] = fmaxf(a, 0.f);
  __syncthreads();
  a = fc2b[j];
  for (int k = 0; k < 128; ++k) a += h1[k] * fc2w[j * 128 + k];
  h2[j] = a;
  __syncthreads();
  a = srb[j];
  for (int k = 0; k < 128; ++k) a += h2[k] * srw[j * 128 + k];
  s1[j] = fmaxf(a, 0.f);
  __syncthreads();
  a = seb[j];
  for (int k = 0; k < 128; ++k) a += s1[k] * sew[j * 128 + k];
  sig[img * 128 + j] = 1.f / (1.f + expf(-a));
}

// ---------------------------------------------------------------------------
// WMMA fragment loaders (contiguous LDS runs -> ds_load_b128).
// ---------------------------------------------------------------------------
__device__ __forceinline__ v16bf ld_afrag(const __bf16* row, int lhalf) {
  const v8bf lo = *(const v8bf*)(row + (lhalf << 3));
  const v8bf hi = *(const v8bf*)(row + 16 + (lhalf << 3));
  return __builtin_shufflevector(lo, hi, 0, 1, 2, 3, 4, 5, 6, 7,
                                 8, 9, 10, 11, 12, 13, 14, 15);
}

__device__ __forceinline__ void wmma_step(const __bf16 (*As)[40], const __bf16 (*Bst)[40],
                                          int wr, int wc, int lmod, int lhalf,
                                          v8f acc[2][2]) {
  v16bf af[2], bb[2];
  #pragma unroll
  for (int mt = 0; mt < 2; ++mt)
    af[mt] = ld_afrag(As[wr + (mt << 4) + lmod], lhalf);
  #pragma unroll
  for (int nt = 0; nt < 2; ++nt)
    bb[nt] = ld_afrag(Bst[wc + (nt << 4) + lmod], lhalf);
  #pragma unroll
  for (int mt = 0; mt < 2; ++mt)
    #pragma unroll
    for (int nt = 0; nt < 2; ++nt)
      acc[mt][nt] = __builtin_amdgcn_wmma_f32_16x16x32_bf16(
          false, af[mt], false, bb[nt], (short)0, acc[mt][nt], false, false);
}

// ---------------------------------------------------------------------------
// Implicit-GEMM conv / deconv, double-buffered:
// A tile (bf16 weights, Mpad x Kpad, zero padded) is DMA'd to LDS by the TDM
// one K-step ahead (s_wait_tensorcnt 1 keeps the newest transfer in flight);
// B tile is an im2col gather staged into the alternate buffer concurrently.
// Block: 128 threads (4 wave32), 64x64 output tile, BK=32.
// ---------------------------------------------------------------------------
struct ConvGemm {
  const __bf16* W;   // Mpad x Kp, zero padded
  const float*  X;
  const float*  bias;
  float* Y;
  int M, N, K, Kp, Mpad, Cin;
  int H, Wi, Ho, Wo;
  int kh, kw, stride, pad;
  long xImgStride, yImgStride;
};

template<int DECONV>
__global__ __launch_bounds__(128)
void k_conv_gemm(ConvGemm p) {
  __shared__ alignas(16) __bf16 As[2][64][40];
  __shared__ alignas(16) __bf16 Bst[2][64][40];
  const float* X = p.X + (long)blockIdx.z * p.xImgStride;
  float* Y = p.Y + (long)blockIdx.z * p.yImgStride;
  const int m0 = blockIdx.y << 6;
  const int n0 = blockIdx.x << 6;
  const int tid = threadIdx.x;
  const int lane = tid & 31;
  const int wave = tid >> 5;
  const int wr = (wave >> 1) << 5;
  const int wc = (wave & 1) << 5;
  const int lmod = lane & 15;
  const int lhalf = lane >> 4;
  const int ksz = p.kh * p.kw;

  // Per-thread B-staging coordinates: fixed output column, 16 consecutive k.
  const int nc = tid & 63;
  const int kr0 = (tid >> 6) << 4;  // 0 or 16
  const int gn = n0 + nc;           // N need not be a multiple of 64 (480); cols >= N are not stored
  const int oy = gn / p.Wo;
  const int ox = gn - oy * p.Wo;
  const int oyS = DECONV ? (oy - 1) : (oy * p.stride - p.pad);
  const int oxS = DECONV ? (ox - 1) : (ox * p.stride - p.pad);

  v8f acc[2][2] = {};
  const int KT = (p.K + 31) >> 5;
  const unsigned long long wbase = (unsigned long long)(size_t)p.W;

  // TDM issue of one A tile into an LDS buffer (ISA 08_async_tensor D#).
  auto tdmA = [&](int kbase, const __bf16* ldsDst) {
    unsigned long long ga =
        wbase + (((unsigned long long)(unsigned)m0 * (unsigned)p.Kp + (unsigned)kbase) << 1);
    u32x4 g0;
    g0[0] = 1u;                                 // count=1, user descriptor
    g0[1] = (unsigned)(size_t)ldsDst;           // lds_addr
    g0[2] = (unsigned)ga;                       // global_addr[31:0]
    g0[3] = (unsigned)(ga >> 32) | (2u << 30);  // global_addr[56:32] | type=2
    const unsigned td0 = (unsigned)(p.Kp - kbase);   // >= 32 (zero-padded)
    const unsigned td1 = (unsigned)(p.Mpad - m0);    // >= 64 (zero-padded)
    i32x8 g1;
    // data_size=2B, pad_enable, pad_interval=16 dwords, pad_amount=4 dwords
    // -> LDS row pitch = 64B data + 16B pad = 80B = As row stride.
    g1[0] = (int)((1u << 16) | (1u << 20) | (3u << 22) | (3u << 25));
    g1[1] = (int)((td0 & 0xFFFFu) << 16);            // tensor_dim0[15:0]
    g1[2] = (int)((td0 >> 16) | ((td1 & 0xFFFFu) << 16));
    g1[3] = (int)((td1 >> 16) | (32u << 16));        // tile_dim0 = 32
    g1[4] = 64;                                      // tile_dim1 = 64, tile_dim2 = 0
    g1[5] = p.Kp;                                    // tensor_dim0_stride (elements)
    g1[6] = 0;
    g1[7] = 0;
    const i32x4 gz4 = {0, 0, 0, 0};
    const i32x8 gz8 = {0, 0, 0, 0, 0, 0, 0, 0};
    __builtin_amdgcn_tensor_load_to_lds(g0, g1, gz4, gz4, gz8, 0);
  };

  // B tile: im2col gather, stored transposed (n-major, k contiguous).
  // Clamped addressing + mask multiply: unconditional loads, no exec branches.
  auto stageB = [&](int kbase, __bf16 (*B)[40]) {
    int kk = kbase + kr0;
    int c = kk / ksz;
    int rr = kk - c * ksz;
    int ky = rr / p.kw;
    int kx = rr - ky * p.kw;
    __bf16 tmp[16];
    #pragma unroll
    for (int i = 0; i < 16; ++i) {
      int ok, iy, ix;
      if (DECONV) {
        int ty = oyS + ky, tx = oxS + kx;
        iy = ty >> 1; ix = tx >> 1;
        ok = (kk + i < p.K) & (ty >= 0) & (tx >= 0) &
             ((ty & 1) == 0) & ((tx & 1) == 0) & (iy < p.H) & (ix < p.Wi);
      } else {
        iy = oyS + ky; ix = oxS + kx;
        ok = (kk + i < p.K) & (iy >= 0) & (iy < p.H) & (ix >= 0) & (ix < p.Wi);
      }
      int cc  = min(c, p.Cin - 1);
      int iyc = min(max(iy, 0), p.H - 1);
      int ixc = min(max(ix, 0), p.Wi - 1);
      float v = X[(long)cc * p.H * p.Wi + (long)iyc * p.Wi + ixc] * (ok ? 1.f : 0.f);
      tmp[i] = (__bf16)v;
      if (++kx == p.kw) { kx = 0; if (++ky == p.kh) { ky = 0; ++c; } }
    }
    *(v8bf*)&B[nc][kr0]     = *(const v8bf*)&tmp[0];
    *(v8bf*)&B[nc][kr0 + 8] = *(const v8bf*)&tmp[8];
  };

  // Prologue: tile 0 into buffer 0.
  if (wave == 0) tdmA(0, &As[0][0][0]);
  stageB(0, Bst[0]);

  for (int kt = 0; kt < KT; ++kt) {
    const int cur = kt & 1;
    const int nxt = cur ^ 1;
    // Issue next tile into the alternate buffer (safe: the barrier ending the
    // previous iteration guarantees nobody still reads it).
    if (kt + 1 < KT) {
      if (wave == 0) tdmA((kt + 1) << 5, &As[nxt][0][0]);
      stageB((kt + 1) << 5, Bst[nxt]);
    }
    // Ensure tile kt's TDM landed (allow the newest transfer to stay in flight).
    if (wave == 0) {
      if (kt + 1 < KT) __builtin_amdgcn_s_wait_tensorcnt(1);
      else             __builtin_amdgcn_s_wait_tensorcnt(0);
    }
    __syncthreads();
    wmma_step(As[cur], Bst[cur], wr, wc, lmod, lhalf, acc);
    __syncthreads();
  }

  // ---- epilogue: C/D layout n=lane%16, m = r + 8*(lane/16)
  #pragma unroll
  for (int mt = 0; mt < 2; ++mt) {
    #pragma unroll
    for (int nt = 0; nt < 2; ++nt) {
      int col = n0 + wc + (nt << 4) + lmod;
      #pragma unroll
      for (int r = 0; r < 8; ++r) {
        int row = m0 + wr + (mt << 4) + r + (lhalf << 3);
        if (row < p.M && col < p.N) {
          float v = acc[mt][nt][r];
          if (p.bias) v += p.bias[row];
          Y[(long)row * p.N + col] = v;
        }
      }
    }
  }
}

// ---------------------------------------------------------------------------
// Cost-volume GEMM: M[b,h,w,v] = sum_c left[b,c,h,w] * right[b,c,h,v]
// Per slice z=(b,h): A[m,k]=L[k*7680+m], B[k,n]=R[k*7680+n]; M=N=160, K=64.
// fp32 source needs conversion, so staged manually (clamped + mask-select).
// ---------------------------------------------------------------------------
__global__ __launch_bounds__(128)
void k_cost_gemm(const float* __restrict__ feat2, float* __restrict__ Mout) {
  __shared__ alignas(16) __bf16 As[64][40];
  __shared__ alignas(16) __bf16 Bst[64][40];
  const int z = blockIdx.z;
  const int b = z / 48;
  const int h = z - b * 48;
  const float* L = feat2 + (long)(2 * b) * 64 * 7680 + (long)h * 160;
  const float* R = feat2 + (long)(2 * b + 1) * 64 * 7680 + (long)h * 160;
  float* Y = Mout + (long)z * 25600;
  const int m0 = blockIdx.y << 6;
  const int n0 = blockIdx.x << 6;
  const int tid = threadIdx.x;
  const int lane = tid & 31;
  const int wave = tid >> 5;
  const int wr = (wave >> 1) << 5;
  const int wc = (wave & 1) << 5;
  const int lmod = lane & 15;
  const int lhalf = lane >> 4;
  const int nc = tid & 63;
  const int kr0 = (tid >> 6) << 4;
  const int gn = n0 + nc;
  const int gnc = min(gn, 159);
  const float nmask = (gn < 160) ? 1.f : 0.f;
  v8f acc[2][2] = {};
  __builtin_prefetch(L + (long)32 * 7680 + tid, 0, 1);  // next K slab
  for (int kt = 0; kt < 2; ++kt) {
    const int kbase = kt << 5;
    // A: consecutive lanes -> consecutive m (coalesced); clamped addressing
    #pragma unroll
    for (int i = 0; i < 16; ++i) {
      int idx = (i << 7) + tid;
      int r = idx & 63;        // m within tile
      int cK = idx >> 6;       // k within tile (0..31)
      int gm = m0 + r;
      int gmc = min(gm, 159);
      float mk = (gm < 160) ? 1.f : 0.f;
      As[r][cK] = (__bf16)(L[(long)(kbase + cK) * 7680 + gmc] * mk);
    }
    // B: fixed column per thread, 16 consecutive k, vector LDS store
    {
      __bf16 tmp[16];
      #pragma unroll
      for (int i = 0; i < 16; ++i)
        tmp[i] = (__bf16)(R[(long)(kbase + kr0 + i) * 7680 + gnc] * nmask);
      *(v8bf*)&Bst[nc][kr0]     = *(const v8bf*)&tmp[0];
      *(v8bf*)&Bst[nc][kr0 + 8] = *(const v8bf*)&tmp[8];
    }
    __syncthreads();
    wmma_step(As, Bst, wr, wc, lmod, lhalf, acc);
    __syncthreads();
  }
  #pragma unroll
  for (int mt = 0; mt < 2; ++mt) {
    #pragma unroll
    for (int nt = 0; nt < 2; ++nt) {
      int col = n0 + wc + (nt << 4) + lmod;
      #pragma unroll
      for (int r = 0; r < 8; ++r) {
        int row = m0 + wr + (mt << 4) + r + (lhalf << 3);
        if (row < 160 && col < 160)
          Y[(long)row * 160 + col] = acc[mt][nt][r];
      }
    }
  }
}

// ---------------------------------------------------------------------------
// GroupNorm stats (mean, rstd) per (image, group)
// ---------------------------------------------------------------------------
__global__ __launch_bounds__(256)
void k_gn_stats(const float* __restrict__ X, float* __restrict__ stats,
                int C, long HW, int groups) {
  const int blk = blockIdx.x;
  const int img = blk / groups;
  const int g = blk - img * groups;
  const int cpg = C / groups;
  const long n = (long)cpg * HW;
  const long base = (long)img * C * HW + (long)g * cpg * HW;
  float s = 0.f, q = 0.f;
  for (long i = threadIdx.x; i < n; i += 256) {
    float v = X[base + i];
    s += v; q += v * v;
  }
  __shared__ float rs[256], rq[256];
  rs[threadIdx.x] = s; rq[threadIdx.x] = q;
  __syncthreads();
  for (int off = 128; off > 0; off >>= 1) {
    if (threadIdx.x < off) {
      rs[threadIdx.x] += rs[threadIdx.x + off];
      rq[threadIdx.x] += rq[threadIdx.x + off];
    }
    __syncthreads();
  }
  if (threadIdx.x == 0) {
    float mu = rs[0] / (float)n;
    float var = rq[0] / (float)n - mu * mu;
    stats[2 * blk] = mu;
    stats[2 * blk + 1] = rsqrtf(fmaxf(var, 0.f) + 1e-5f);
  }
}

__global__ void k_gn_apply(float* __restrict__ X, const float* __restrict__ stats,
                           const float* __restrict__ gam, const float* __restrict__ bet,
                           const float* __restrict__ se, int C, long HW, int groups,
                           int relu, long total) {
  long i = (long)blockIdx.x * blockDim.x + threadIdx.x;
  if (i >= total) return;
  long cHW = (long)C * HW;
  int img = (int)(i / cHW);
  int c = (int)((i - (long)img * cHW) / HW);
  int grp = c / (C / groups);
  float mu = stats[2 * (img * groups + grp)];
  float rstd = stats[2 * (img * groups + grp) + 1];
  float v = (X[i] - mu) * rstd * gam[c] + bet[c];
  if (relu) v = fmaxf(v, 0.f);
  if (se) v *= se[img * C + c];
  X[i] = v;
}

// y = relu(bn_eval(x) + add)
__global__ void k_affine_add_relu(const float* __restrict__ X, const float* __restrict__ gam,
                                  const float* __restrict__ bet, const float* __restrict__ add,
                                  float* __restrict__ Y, int C, long HW, long total) {
  long i = (long)blockIdx.x * blockDim.x + threadIdx.x;
  if (i >= total) return;
  int c = (int)((i / HW) % C);
  float v = X[i] * (gam[c] * BN_SCALE) + bet[c] + add[i];
  Y[i] = fmaxf(v, 0.f);
}

// ---------------------------------------------------------------------------
// Depth-hypothesis bilinear gather from correlation matrices.
// ---------------------------------------------------------------------------
__global__ void k_cost_interp(const float* __restrict__ Mmat, const float* __restrict__ calib,
                              float* __restrict__ cost, long total) {
  long i = (long)blockIdx.x * blockDim.x + threadIdx.x;
  if (i >= total) return;
  int x = (int)(i % 160);
  int h = (int)((i / 160) % 48);
  int d = (int)((i / 7680) % 112);
  int b = (int)(i / (7680L * 112));
  float dsv = 2.0f + 0.5f * (float)d;
  float off = calib[b] / 8.0f / dsv;
  float yy = (float)x - off;
  float lm = (yy >= 0.f && yy <= 159.f) ? 1.f : 0.f;
  float y0 = floorf(yy);
  float t = yy - y0;
  int y0i = (int)y0;
  int y1i = y0i + 1;
  float m0 = (y0i >= 0 && y0i <= 159) ? 1.f : 0.f;
  float m1 = (y1i >= 0 && y1i <= 159) ? 1.f : 0.f;
  int y0c = min(max(y0i, 0), 159);
  int y1c = min(max(y1i, 0), 159);
  const float* Mrow = Mmat + ((long)(b * 48 + h) * 160 + x) * 160;
  float v = ((1.f - t) * m0) * Mrow[y0c] + (t * m1) * Mrow[y1c];
  cost[i] = v * lm;
}

// ---------------------------------------------------------------------------
// Host orchestration
// ---------------------------------------------------------------------------
extern "C" void kernel_launch(void* const* d_in, const int* in_sizes, int n_in,
                              void* d_out, int out_size, void* d_ws, size_t ws_size,
                              hipStream_t stream) {
  (void)in_sizes; (void)n_in; (void)out_size; (void)ws_size;
  auto F = [&](int i) { return (const float*)d_in[i]; };

  const float* x        = F(0);
  const float* mlp      = F(1);
  const float* calib    = F(2);
  const float* out_b    = F(3);
  const float* out_w    = F(4);
  const float* bn_b     = F(5);
  const float* bn_g     = F(6);
  const float* fc1_b    = F(7);
  const float* fc1_w    = F(8);
  const float* fc2_b    = F(9);
  const float* fc2_w    = F(10);
  const float* feat_b   = F(11);
  const float* feat_w   = F(12);
  const float* gn_b     = F(13);
  const float* gn_g     = F(14);
  const float* reduce_b = F(15);
  const float* reduce_w = F(16);
  const float* se_exp_b = F(17);
  const float* se_exp_w = F(18);
  const float* se_red_b = F(19);
  const float* se_red_w = F(20);

  auto MP = [](int M) { return (M + 63) & ~63; };
  auto KP = [](int K) { return (K + 31) & ~31; };

  // bump allocator over d_ws
  char* wsb = (char*)d_ws;
  size_t off = 0;
  auto alloc = [&](size_t bytes) -> void* {
    off = (off + 255) & ~(size_t)255;
    void* pp = wsb + off;
    off += bytes;
    return pp;
  };
  auto allocF = [&](long n) { return (float*)alloc((size_t)n * 4); };
  auto allocW = [&](int M, int K) { return (__bf16*)alloc((size_t)MP(M) * KP(K) * 2); };

  float* sig   = allocF(4 * 128);
  float* stats = allocF(64);

  __bf16* w_reduce = allocW(128, 1152);
  __bf16* w_feat   = allocW(64, 128);
  __bf16* w_out    = allocW(112, 112);
  struct UnetW { __bf16 *c1, *c2, *c3, *c4, *c5t, *c6t, *r1, *r2; };
  UnetW uw[2];
  for (int u = 0; u < 2; ++u) {
    uw[u].c1  = allocW(224, 1008);
    uw[u].c2  = allocW(224, 2016);
    uw[u].c3  = allocW(448, 2016);
    uw[u].c4  = allocW(448, 4032);
    uw[u].c5t = allocW(224, 4032);
    uw[u].c6t = allocW(112, 2016);
    uw[u].r1  = allocW(112, 112);
    uw[u].r2  = allocW(224, 224);
  }

  float* bufA  = allocF((long)4 * 128 * 7680);
  float* feat2 = allocF((long)4 * 64 * 7680);
  float* Mmat  = allocF((long)96 * 160 * 160);
  float* cost  = allocF((long)2 * 112 * 7680);
  float* u_c1  = allocF((long)2 * 224 * 1920);
  float* u_c2  = allocF((long)2 * 224 * 1920);
  float* u_c3  = allocF((long)2 * 448 * 480);
  float* u_c4  = allocF((long)2 * 448 * 480);
  float* u_r2  = allocF((long)2 * 224 * 1920);
  float* u_c5  = allocF((long)2 * 224 * 1920);
  float* c5act = allocF((long)2 * 224 * 1920);
  float* u_r1  = allocF((long)2 * 112 * 7680);
  float* u_c6  = allocF((long)2 * 112 * 7680);
  float* u1out = allocF((long)2 * 112 * 7680);
  float* u2out = allocF((long)2 * 112 * 7680);

  auto cvt = [&](const float* src, __bf16* dst, int M, int K) {
    long total = (long)MP(M) * KP(K);
    k_cvt_pad_bf16<<<dim3((unsigned)((total + 255) / 256)), dim3(256), 0, stream>>>(
        src, dst, M, K, KP(K), total);
  };
  auto cvtdec = [&](const float* src, __bf16* dst, int Ci, int Co) {
    long total = (long)MP(Co) * KP(Ci * 9);
    k_cvt_deconv_pad_bf16<<<dim3((unsigned)((total + 255) / 256)), dim3(256), 0, stream>>>(
        src, dst, Ci, Co, KP(Ci * 9), total);
  };

  cvt(reduce_w, w_reduce, 128, 1152);
  cvt(feat_w, w_feat, 64, 128);
  cvt(out_w, w_out, 112, 112);
  for (int u = 0; u < 2; ++u) {
    int b0 = 21 + 24 * u;
    cvt(F(b0 + 2),  uw[u].c1, 224, 1008);
    cvt(F(b0 + 5),  uw[u].c2, 224, 2016);
    cvt(F(b0 + 8),  uw[u].c3, 448, 2016);
    cvt(F(b0 + 11), uw[u].c4, 448, 4032);
    cvtdec(F(b0 + 14), uw[u].c5t, 448, 224);
    cvtdec(F(b0 + 17), uw[u].c6t, 224, 112);
    cvt(F(b0 + 20), uw[u].r1, 112, 112);
    cvt(F(b0 + 23), uw[u].r2, 224, 224);
  }

  // SE path -> per (img,channel) sigmoid gate
  k_se<<<4, 128, 0, stream>>>(mlp, bn_g, bn_b, fc1_w, fc1_b, fc2_w, fc2_b,
                              se_red_w, se_red_b, se_exp_w, se_exp_b, sig);

  auto conv = [&](const __bf16* Wb, const float* Xp, const float* bias, float* Yp,
                  int M, int Cin, int H, int Wi, int kh, int strd, int pad,
                  int nimg, bool deconv) {
    ConvGemm p;
    p.W = Wb; p.X = Xp; p.bias = bias; p.Y = Yp;
    p.H = H; p.Wi = Wi; p.Cin = Cin;
    p.kh = kh; p.kw = kh; p.stride = strd; p.pad = pad;
    if (deconv) { p.Ho = 2 * H; p.Wo = 2 * Wi; }
    else { p.Ho = (H + 2 * pad - kh) / strd + 1; p.Wo = (Wi + 2 * pad - kh) / strd + 1; }
    p.M = M; p.N = p.Ho * p.Wo; p.K = Cin * kh * kh;
    p.Kp = KP(p.K); p.Mpad = MP(M);
    p.xImgStride = (long)Cin * H * Wi;
    p.yImgStride = (long)M * p.N;
    dim3 grid((unsigned)((p.N + 63) / 64), (unsigned)((M + 63) / 64), (unsigned)nimg);
    if (deconv) k_conv_gemm<1><<<grid, 128, 0, stream>>>(p);
    else        k_conv_gemm<0><<<grid, 128, 0, stream>>>(p);
  };

  auto gnorm = [&](float* Xp, const float* gam, const float* bet, const float* se,
                   int nimg, int C, long HW, bool relu) {
    k_gn_stats<<<nimg * 2, 256, 0, stream>>>(Xp, stats, C, HW, 2);
    long total = (long)nimg * C * HW;
    k_gn_apply<<<(unsigned)((total + 255) / 256), 256, 0, stream>>>(
        Xp, stats, gam, bet, se, C, HW, 2, relu ? 1 : 0, total);
  };

  // SFN: reduce conv (3x3) -> GN+ReLU+SE gate -> feat 1x1 conv
  conv(w_reduce, x, reduce_b, bufA, 128, 128, 48, 160, 3, 1, 1, 4, false);
  gnorm(bufA, gn_g, gn_b, sig, 4, 128, 7680, true);
  conv(w_feat, bufA, feat_b, feat2, 64, 128, 48, 160, 1, 1, 0, 4, false);

  // Cost volume: per-row correlation GEMMs + depth gather
  k_cost_gemm<<<dim3(3, 3, 96), 128, 0, stream>>>(feat2, Mmat);
  {
    long total = (long)2 * 112 * 48 * 160;
    k_cost_interp<<<(unsigned)((total + 255) / 256), 256, 0, stream>>>(Mmat, calib, cost, total);
  }

  auto run_unet = [&](const float* xin, float* outp, int b0, const UnetW& w) {
    conv(w.c1, xin, nullptr, u_c1, 224, 112, 48, 160, 3, 2, 1, 2, false);
    gnorm(u_c1, F(b0 + 1), F(b0 + 0), nullptr, 2, 224, 1920, true);
    conv(w.c2, u_c1, nullptr, u_c2, 224, 224, 24, 80, 3, 1, 1, 2, false);
    gnorm(u_c2, F(b0 + 4), F(b0 + 3), nullptr, 2, 224, 1920, true);
    conv(w.c3, u_c2, nullptr, u_c3, 448, 224, 24, 80, 3, 2, 1, 2, false);
    gnorm(u_c3, F(b0 + 7), F(b0 + 6), nullptr, 2, 448, 480, true);
    conv(w.c4, u_c3, nullptr, u_c4, 448, 448, 12, 40, 3, 1, 1, 2, false);
    gnorm(u_c4, F(b0 + 10), F(b0 + 9), nullptr, 2, 448, 480, true);
    conv(w.r2, u_c2, nullptr, u_r2, 224, 224, 24, 80, 1, 1, 0, 2, false);
    gnorm(u_r2, F(b0 + 22), F(b0 + 21), nullptr, 2, 224, 1920, false);
    conv(w.c5t, u_c4, nullptr, u_c5, 224, 448, 12, 40, 3, 1, 1, 2, true);
    {
      long total = (long)2 * 224 * 1920;
      k_affine_add_relu<<<(unsigned)((total + 255) / 256), 256, 0, stream>>>(
          u_c5, F(b0 + 13), F(b0 + 12), u_r2, c5act, 224, 1920L, total);
    }
    conv(w.r1, xin, nullptr, u_r1, 112, 112, 48, 160, 1, 1, 0, 2, false);
    gnorm(u_r1, F(b0 + 19), F(b0 + 18), nullptr, 2, 112, 7680, false);
    conv(w.c6t, c5act, nullptr, u_c6, 112, 224, 24, 80, 3, 1, 1, 2, true);
    {
      long total = (long)2 * 112 * 7680;
      k_affine_add_relu<<<(unsigned)((total + 255) / 256), 256, 0, stream>>>(
          u_c6, F(b0 + 16), F(b0 + 15), u_r1, outp, 112, 7680L, total);
    }
  };

  run_unet(cost, u1out, 21, uw[0]);
  run_unet(u1out, u2out, 45, uw[1]);

  // Final 1x1 conv -> d_out
  conv(w_out, u2out, out_b, (float*)d_out, 112, 112, 48, 160, 1, 1, 0, 2, false);
}
